// MultiHeadAttention_88802743812341
// MI455X (gfx1250) — matrix operand order repaired and checked by reading, not compiled, not run on hardware
//
#include <hip/hip_runtime.h>
#include <hip/hip_bf16.h>

typedef unsigned short ushort_t;
typedef __attribute__((ext_vector_type(16))) __bf16 v16bf;
typedef __attribute__((ext_vector_type(8)))  float  v8f;

#define B_  2
#define S_  2048
#define D_  1024
#define H_  16
#define DH_ 64
#define M_  (B_ * S_)          // 4096 rows
#define LN_EPS 1e-5f
#define KB_ 128                // GEMM K-block staged in LDS

#if defined(__has_builtin)
# if __has_builtin(__builtin_amdgcn_global_load_async_to_lds_b128)
#  define HAVE_ASYNC_LDS 1
# endif
#endif

#if defined(HAVE_ASYNC_LDS)
typedef int v4i_ __attribute__((vector_size(16)));
typedef __attribute__((address_space(1))) v4i_ g_v4i;   // global
typedef __attribute__((address_space(3))) v4i_ l_v4i;   // LDS
# if __has_builtin(__builtin_amdgcn_s_wait_asynccnt)
#  define WAIT_ASYNC(n) __builtin_amdgcn_s_wait_asynccnt(n)
# else
#  define WAIT_ASYNC(n) asm volatile("s_wait_asynccnt %0" :: "i"(n) : "memory")
# endif
#else
# define WAIT_ASYNC(n) ((void)0)
#endif

union FragBF {
    v16bf v;
    uint4 q[2];
};

__device__ __forceinline__ ushort_t f2bf(float f) {
    unsigned int u = __float_as_uint(f);
    u += 0x7fffu + ((u >> 16) & 1u);   // round-to-nearest-even
    return (ushort_t)(u >> 16);
}

// Copy one 16B chunk global -> LDS (async on CDNA5, sync fallback otherwise)
__device__ __forceinline__ void copy16_g2l(const ushort_t* g, ushort_t* l) {
#if defined(HAVE_ASYNC_LDS)
    __builtin_amdgcn_global_load_async_to_lds_b128(
        (g_v4i*)(void*)const_cast<ushort_t*>(g), (l_v4i*)(void*)l, 0, 0);
#else
    *(uint4*)l = *(const uint4*)g;
#endif
}

// Stage a 16 x KB_ bf16 A-tile into LDS (256 chunks of 16B, 128 threads x 2)
__device__ __forceinline__ void stage_A(const ushort_t* __restrict__ A,
                                        int m0, int kb, int Kk, ushort_t* As) {
    const int t = threadIdx.x;
    #pragma unroll
    for (int it = 0; it < 2; ++it) {
        int c   = t + it * 128;      // 0..255
        int row = c >> 4;            // 16 chunks per row (KB_*2B / 16B)
        int cc  = c & 15;
        copy16_g2l(A + (size_t)(m0 + row) * Kk + kb * KB_ + cc * 8,
                   As + row * KB_ + cc * 8);
    }
}

// ---------------------------------------------------------------- fp32 -> bf16
__global__ void cvt_bf16_kernel(const float* __restrict__ in, ushort_t* __restrict__ out, int n) {
    int i = blockIdx.x * blockDim.x + threadIdx.x;
    if (i < n) out[i] = f2bf(in[i]);
}

// --------------------------------------------- W[k][n] fp32 -> Wt[n][k] bf16
__global__ void cvt_wt_kernel(const float* __restrict__ W, ushort_t* __restrict__ Wt) {
    int n = blockIdx.x;
    for (int k = threadIdx.x; k < D_; k += blockDim.x)
        Wt[(size_t)n * D_ + k] = f2bf(W[(size_t)k * D_ + n]);
}

// ------------------------------- vh[b*S+s][h*64+d] bf16 -> v_t[b][h][d][s] bf16
__global__ void transpose_v_kernel(const ushort_t* __restrict__ vh, ushort_t* __restrict__ vt) {
    size_t i = (size_t)blockIdx.x * blockDim.x + threadIdx.x;
    if (i >= (size_t)B_ * H_ * DH_ * S_) return;
    int s  = (int)(i % S_);
    int d  = (int)((i / S_) % DH_);
    int h  = (int)((i / ((size_t)S_ * DH_)) % H_);
    int b  = (int)(i / ((size_t)S_ * DH_ * H_));
    vt[i] = vh[((size_t)(b * S_ + s)) * D_ + h * DH_ + d];
}

// -------------------------------------------------------------- bf16 GEMM
// C_bf16[M x N] = scale * (A_bf16[M x K] @ Wt_bf16[N x K]^T + bias)
// block = 128 threads (4 waves), 16x64 block tile; A staged via async LDS
__global__ void gemm_bf16_kernel(const ushort_t* __restrict__ A,
                                 const ushort_t* __restrict__ Wt,
                                 const float* __restrict__ bias,
                                 ushort_t* __restrict__ out,
                                 int Mm, int Nn, int Kk, float scale) {
    __shared__ ushort_t As[2][16 * KB_];
    const int lane = threadIdx.x & 31;
    const int wave = threadIdx.x >> 5;
    const int half = lane >> 4;
    const int lc   = lane & 15;
    const int m0 = blockIdx.x * 16;
    const int n0 = blockIdx.y * 64 + wave * 16;
    const int nkb = Kk / KB_;

    v8f c = {};
    const ushort_t* brow = Wt + (size_t)(n0 + lc) * Kk + 8 * half;

    stage_A(A, m0, 0, Kk, As[0]);
    for (int kb = 0; kb < nkb; ++kb) {
        if (kb + 1 < nkb) {
            stage_A(A, m0, kb + 1, Kk, As[(kb + 1) & 1]);
            WAIT_ASYNC(2);
        } else {
            WAIT_ASYNC(0);
        }
        __syncthreads();
        const ushort_t* as = As[kb & 1] + lc * KB_ + 8 * half;
        #pragma unroll
        for (int k0 = 0; k0 < KB_; k0 += 32) {
            FragBF a, b;
            a.q[0] = *(const uint4*)(as + k0);
            a.q[1] = *(const uint4*)(as + k0 + 16);
            b.q[0] = *(const uint4*)(brow + kb * KB_ + k0);
            b.q[1] = *(const uint4*)(brow + kb * KB_ + k0 + 16);
            c = __builtin_amdgcn_wmma_f32_16x16x32_bf16(false, a.v, false, b.v,
                                                        (short)0, c, false, false);
        }
        __syncthreads();
    }
    const int n = n0 + lc;
    const float bn = bias[n];
    #pragma unroll
    for (int r = 0; r < 8; ++r) {
        int m = m0 + r + 8 * half;
        out[(size_t)m * Nn + n] = f2bf(scale * (c[r] + bn));
    }
}

// --------------------------------- output projection: fp32 out + bias + residual
__global__ void gemm_out_kernel(const ushort_t* __restrict__ A,
                                const ushort_t* __restrict__ Wt,
                                const float* __restrict__ bias,
                                const float* __restrict__ residual,
                                float* __restrict__ y) {
    __shared__ ushort_t As[2][16 * KB_];
    const int lane = threadIdx.x & 31;
    const int wave = threadIdx.x >> 5;
    const int half = lane >> 4;
    const int lc   = lane & 15;
    const int m0 = blockIdx.x * 16;
    const int n0 = blockIdx.y * 64 + wave * 16;
    const int nkb = D_ / KB_;

    v8f c = {};
    const ushort_t* brow = Wt + (size_t)(n0 + lc) * D_ + 8 * half;

    stage_A(A, m0, 0, D_, As[0]);
    for (int kb = 0; kb < nkb; ++kb) {
        if (kb + 1 < nkb) {
            stage_A(A, m0, kb + 1, D_, As[(kb + 1) & 1]);
            WAIT_ASYNC(2);
        } else {
            WAIT_ASYNC(0);
        }
        __syncthreads();
        const ushort_t* as = As[kb & 1] + lc * KB_ + 8 * half;
        #pragma unroll
        for (int k0 = 0; k0 < KB_; k0 += 32) {
            FragBF a, b;
            a.q[0] = *(const uint4*)(as + k0);
            a.q[1] = *(const uint4*)(as + k0 + 16);
            b.q[0] = *(const uint4*)(brow + kb * KB_ + k0);
            b.q[1] = *(const uint4*)(brow + kb * KB_ + k0 + 16);
            c = __builtin_amdgcn_wmma_f32_16x16x32_bf16(false, a.v, false, b.v,
                                                        (short)0, c, false, false);
        }
        __syncthreads();
    }
    const int n = n0 + lc;
    const float bn = bias[n];
    #pragma unroll
    for (int r = 0; r < 8; ++r) {
        int m = m0 + r + 8 * half;
        y[(size_t)m * D_ + n] = c[r] + bn + residual[(size_t)m * D_ + n];
    }
}

// ------------------------------------------------------- fused attention core
// grid (S/16, H, B), block 128 (4 waves)
// dyn LDS: fp32 scores 16x2048 (128KB) + bf16 probs 16x2048 (64KB)
__global__ void attention_kernel(const ushort_t* __restrict__ qh,
                                 const ushort_t* __restrict__ kh,
                                 const ushort_t* __restrict__ vt,
                                 const unsigned char* __restrict__ mask,
                                 float* __restrict__ attn_out,
                                 ushort_t* __restrict__ ao) {
    extern __shared__ char smem[];
    float*    Sc = (float*)smem;                                  // [16][2048]
    ushort_t* Pb = (ushort_t*)(smem + 16 * 2048 * sizeof(float)); // [16][2048]
    __shared__ float red[128];
    __shared__ float rowmax[16];
    __shared__ float rowsum[16];

    const int b  = blockIdx.z;
    const int h  = blockIdx.y;
    const int q0 = blockIdx.x * 16;
    const int lane = threadIdx.x & 31;
    const int wave = threadIdx.x >> 5;
    const int half = lane >> 4;
    const int lc   = lane & 15;

    // ---- phase 1: scores = Q @ K^T (scale pre-folded into Q)
    // Q fragments are kt-invariant: hoist the 4 b128 loads out of the loop.
    const ushort_t* arow = qh + ((size_t)(b * S_ + q0 + lc)) * D_ + h * DH_ + 8 * half;
    FragBF aQ0, aQ1;
    aQ0.q[0] = *(const uint4*)(arow + 0);
    aQ0.q[1] = *(const uint4*)(arow + 16);
    aQ1.q[0] = *(const uint4*)(arow + 32);
    aQ1.q[1] = *(const uint4*)(arow + 48);

    for (int kt = wave; kt < S_ / 16; kt += 4) {
        v8f c = {};
        const int sk = kt * 16 + lc;
        const ushort_t* brow = kh + ((size_t)(b * S_ + sk)) * D_ + h * DH_ + 8 * half;
        FragBF b0, b1;
        b0.q[0] = *(const uint4*)(brow + 0);
        b0.q[1] = *(const uint4*)(brow + 16);
        b1.q[0] = *(const uint4*)(brow + 32);
        b1.q[1] = *(const uint4*)(brow + 48);
        c = __builtin_amdgcn_wmma_f32_16x16x32_bf16(false, aQ0.v, false, b0.v,
                                                    (short)0, c, false, false);
        c = __builtin_amdgcn_wmma_f32_16x16x32_bf16(false, aQ1.v, false, b1.v,
                                                    (short)0, c, false, false);
        #pragma unroll
        for (int r = 0; r < 8; ++r)
            Sc[(r + 8 * half) * S_ + kt * 16 + lc] = c[r];
    }
    __syncthreads();

    // ---- phase 2: masked softmax over 2048, write fp32 attn, bf16 probs
    const int row = threadIdx.x >> 3;    // 0..15
    const int sub = threadIdx.x & 7;     // 8 threads per row
    const int base = sub * (S_ / 8);
    const unsigned char* mrow = mask + ((size_t)b * S_ + q0 + row) * S_;
    float* srow = Sc + row * S_;

    float mx = -INFINITY;
    for (int i = 0; i < S_ / 8; ++i) {
        int j = base + i;
        float s = srow[j];
        if (mrow[j]) s = -INFINITY;
        srow[j] = s;
        mx = fmaxf(mx, s);
    }
    red[threadIdx.x] = mx;
    __syncthreads();
    if (sub == 0) {
        float m2 = red[threadIdx.x];
        #pragma unroll
        for (int j = 1; j < 8; ++j) m2 = fmaxf(m2, red[threadIdx.x + j]);
        rowmax[row] = m2;
    }
    __syncthreads();
    const float rm = rowmax[row];

    float sum = 0.f;
    for (int i = 0; i < S_ / 8; ++i) {
        int j = base + i;
        float e = __expf(srow[j] - rm);
        srow[j] = e;
        sum += e;
    }
    red[threadIdx.x] = sum;
    __syncthreads();
    if (sub == 0) {
        float t = red[threadIdx.x];
        #pragma unroll
        for (int j = 1; j < 8; ++j) t += red[threadIdx.x + j];
        rowsum[row] = t;
    }
    __syncthreads();
    const float inv = 1.0f / rowsum[row];

    float* aout = attn_out + (((size_t)(b * H_ + h) * S_) + q0 + row) * S_;
    for (int i = 0; i < S_ / 8; ++i) {
        int j = base + i;
        float p = srow[j] * inv;
        aout[j] = p;
        Pb[row * S_ + j] = f2bf(p);
    }
    __syncthreads();

    // ---- phase 3: out = P @ V  (V transposed per head: vt[b][h][d][s])
    const int d0 = wave * 16;
    v8f c = {};
    const ushort_t* prow = Pb + lc * S_ + 8 * half;
    const ushort_t* brow = vt + ((size_t)((b * H_ + h) * DH_ + d0 + lc)) * S_ + 8 * half;
    for (int k0 = 0; k0 < S_; k0 += 32) {
        FragBF a, bb;
        a.q[0]  = *(const uint4*)(prow + k0);
        a.q[1]  = *(const uint4*)(prow + k0 + 16);
        bb.q[0] = *(const uint4*)(brow + k0);
        bb.q[1] = *(const uint4*)(brow + k0 + 16);
        c = __builtin_amdgcn_wmma_f32_16x16x32_bf16(false, a.v, false, bb.v,
                                                    (short)0, c, false, false);
    }
    #pragma unroll
    for (int r = 0; r < 8; ++r) {
        int m = b * S_ + q0 + r + 8 * half;
        ao[(size_t)m * D_ + h * DH_ + d0 + lc] = f2bf(c[r]);
    }
}

// ------------------------------------------------------------------ LayerNorm
__global__ void layernorm_kernel(const float* __restrict__ y,
                                 const float* __restrict__ gamma,
                                 const float* __restrict__ beta,
                                 float* __restrict__ out) {
    const int m = blockIdx.x;
    const int t = threadIdx.x;
    __shared__ float s1[256], s2[256];
    float a = 0.f, b = 0.f;
    for (int i = t; i < D_; i += 256) {
        float x = y[(size_t)m * D_ + i];
        a += x; b += x * x;
    }
    s1[t] = a; s2[t] = b;
    __syncthreads();
    for (int off = 128; off > 0; off >>= 1) {
        if (t < off) { s1[t] += s1[t + off]; s2[t] += s2[t + off]; }
        __syncthreads();
    }
    const float mu  = s1[0] * (1.0f / D_);
    const float var = s2[0] * (1.0f / D_) - mu * mu;
    const float inv = rsqrtf(var + LN_EPS);
    for (int i = t; i < D_; i += 256) {
        float x = y[(size_t)m * D_ + i];
        out[(size_t)m * D_ + i] = (x - mu) * inv * gamma[i] + beta[i];
    }
}

extern "C" void kernel_launch(void* const* d_in, const int* in_sizes, int n_in,
                              void* d_out, int out_size, void* d_ws, size_t ws_size,
                              hipStream_t stream) {
    (void)in_sizes; (void)n_in; (void)out_size; (void)ws_size;

    const float* q    = (const float*)d_in[0];
    const float* k    = (const float*)d_in[1];
    const float* v    = (const float*)d_in[2];
    const unsigned char* mask = (const unsigned char*)d_in[3];
    const float* Wq   = (const float*)d_in[4];
    const float* bq   = (const float*)d_in[5];
    const float* Wk   = (const float*)d_in[6];
    const float* bk   = (const float*)d_in[7];
    const float* Wv   = (const float*)d_in[8];
    const float* bv   = (const float*)d_in[9];
    const float* Wo   = (const float*)d_in[10];
    const float* bo   = (const float*)d_in[11];
    const float* lng  = (const float*)d_in[12];
    const float* lnb  = (const float*)d_in[13];

    float* out_final = (float*)d_out;                       // [B,S,D] fp32
    float* attn_out  = out_final + (size_t)M_ * D_;         // [B,H,S,S] fp32

    // ---- workspace carve-up (all offsets 16B aligned)
    char* ws = (char*)d_ws;
    const size_t MD2 = (size_t)M_ * D_ * 2;   // 8 MB (bf16 M x D)
    const size_t DD2 = (size_t)D_ * D_ * 2;   // 2 MB (bf16 D x D)
    ushort_t* qb   = (ushort_t*)(ws);
    ushort_t* kb   = (ushort_t*)(ws + MD2);
    ushort_t* vb   = (ushort_t*)(ws + 2 * MD2);
    ushort_t* Wqt  = (ushort_t*)(ws + 3 * MD2);
    ushort_t* Wkt  = (ushort_t*)(ws + 3 * MD2 + DD2);
    ushort_t* Wvt  = (ushort_t*)(ws + 3 * MD2 + 2 * DD2);
    ushort_t* Wot  = (ushort_t*)(ws + 3 * MD2 + 3 * DD2);
    ushort_t* qhb  = (ushort_t*)(ws + 3 * MD2 + 4 * DD2);
    ushort_t* khb  = (ushort_t*)(ws + 4 * MD2 + 4 * DD2);
    ushort_t* vhb  = (ushort_t*)(ws + 5 * MD2 + 4 * DD2);
    ushort_t* vtb  = (ushort_t*)(ws + 6 * MD2 + 4 * DD2);
    ushort_t* aob  = (ushort_t*)(ws + 7 * MD2 + 4 * DD2);
    float*    ybuf = (float*)   (ws + 8 * MD2 + 4 * DD2);   // M x D fp32

    const int nMD = M_ * D_;
    dim3 cvtg((nMD + 255) / 256);
    cvt_bf16_kernel<<<cvtg, 256, 0, stream>>>(q, qb, nMD);
    cvt_bf16_kernel<<<cvtg, 256, 0, stream>>>(k, kb, nMD);
    cvt_bf16_kernel<<<cvtg, 256, 0, stream>>>(v, vb, nMD);

    cvt_wt_kernel<<<D_, 256, 0, stream>>>(Wq, Wqt);
    cvt_wt_kernel<<<D_, 256, 0, stream>>>(Wk, Wkt);
    cvt_wt_kernel<<<D_, 256, 0, stream>>>(Wv, Wvt);
    cvt_wt_kernel<<<D_, 256, 0, stream>>>(Wo, Wot);

    const float scale = 0.125f;   // 1/sqrt(64), folded into Q
    dim3 gg(M_ / 16, D_ / 64);
    gemm_bf16_kernel<<<gg, 128, 0, stream>>>(qb, Wqt, bq, qhb, M_, D_, D_, scale);
    gemm_bf16_kernel<<<gg, 128, 0, stream>>>(kb, Wkt, bk, khb, M_, D_, D_, 1.0f);
    gemm_bf16_kernel<<<gg, 128, 0, stream>>>(vb, Wvt, bv, vhb, M_, D_, D_, 1.0f);

    dim3 tg((unsigned)(((size_t)B_ * H_ * DH_ * S_ + 255) / 256));
    transpose_v_kernel<<<tg, 256, 0, stream>>>(vhb, vtb);

    const int smem_bytes = 16 * S_ * (int)sizeof(float) + 16 * S_ * (int)sizeof(ushort_t);
    (void)hipFuncSetAttribute(reinterpret_cast<const void*>(attention_kernel),
                              hipFuncAttributeMaxDynamicSharedMemorySize, smem_bytes);
    dim3 ag(S_ / 16, H_, B_);
    attention_kernel<<<ag, 128, smem_bytes, stream>>>(qhb, khb, vtb, mask, attn_out, aob);

    gemm_out_kernel<<<gg, 128, 0, stream>>>(aob, Wot, bo, q, ybuf);

    layernorm_kernel<<<M_, 256, 0, stream>>>(ybuf, lng, lnb, out_final);
}
